// MoleculePredictor_29274497089896
// MI455X (gfx1250) — matrix operand order repaired and checked by reading, not compiled, not run on hardware
//
#include <hip/hip_runtime.h>
#include <hip/hip_bf16.h>

typedef __attribute__((ext_vector_type(16))) _Float16 v16h;
typedef __attribute__((ext_vector_type(8)))  float    v8f;

union Frag16 { v16h v; uint4 q[2]; };

__device__ __forceinline__ float sp_half(float x) {   // softplus beta=0.5, thr=14
  float bx = 0.5f * x;
  return (bx > 14.f) ? x : 2.0f * __logf(1.f + __expf(bx));
}
__device__ __forceinline__ float sp_one(float x) {    // softplus beta=1, thr=20
  return (x > 20.f) ? x : __logf(1.f + __expf(x));
}

// ---------------------------------------------------------------------------
// WMMA GEMM:  Y[M, NT*16] = act( X[M,K](f16) @ Wt[NT*16,K](f16)^T + b )
// Wt pre-transposed (row n = output channel, contiguous K), staged in LDS.
// 4 waves/block; each wave owns TWO 16-row M-tiles (32 rows) so every B
// fragment loaded from LDS feeds two WMMAs. B fragments are loaded in batches
// of 4 (8x ds_load_b128) before the WMMA chain so one wait covers the batch.
// Tail rows are handled by clamping the load row (branch-free inner loop);
// stores are guarded in the epilogue.
// ---------------------------------------------------------------------------
template<int NT>
__global__ void __launch_bounds__(128, 1)
k_gemm(const _Float16* __restrict__ X, int ldx,
       const _Float16* __restrict__ Wt,
       const float* __restrict__ bias,
       int M, int K, int act,
       const float* __restrict__ resid,
       float* __restrict__ outF32,
       _Float16* __restrict__ outH1, int ldh1,
       _Float16* __restrict__ outH2, int ldh2)
{
  extern __shared__ __align__(16) char smem_raw[];
  _Float16* lds = (_Float16*)smem_raw;

  const int tid = threadIdx.x;
  // cooperative load of the whole Wt tile (NT*16 x K halves) into LDS
  {
    const int nq = NT * 16 * K / 8;
    const uint4* s = (const uint4*)Wt;
    uint4* d = (uint4*)lds;
    for (int i = tid; i < nq; i += blockDim.x) d[i] = s[i];
  }
  __syncthreads();

  const int wave = tid >> 5;
  const int lane = tid & 31;
  const int l16  = lane & 15;
  const int hi   = lane >> 4;     // half-wave select
  const int kA   = hi * 8;        // A-fragment K base (ISA 16-bit A layout)
  const int kB   = hi * 16;       // unused; B fragments load at kA, same k order as A
  const int mtiles2 = (M + 31) >> 5;          // 32-row super-tiles

  for (int mt = blockIdx.x * 4 + wave; mt < mtiles2; mt += gridDim.x * 4) {
    const int m0 = mt << 5;
    // branch-free tail: clamp load rows; epilogue guards stores
    int r0 = m0 + l16;       if (r0 > M - 1) r0 = M - 1;
    int r1 = m0 + 16 + l16;  if (r1 > M - 1) r1 = M - 1;
    const _Float16* x0 = X + (size_t)r0 * ldx;
    const _Float16* x1 = X + (size_t)r1 * ldx;

    v8f zero = {};
    v8f acc[2][NT];
#pragma unroll
    for (int t = 0; t < NT; ++t) { acc[0][t] = zero; acc[1][t] = zero; }

    for (int kk = 0; kk < K; kk += 32) {
      Frag16 a0, a1;
      a0.q[0] = *(const uint4*)(x0 + kk + kA);
      a0.q[1] = *(const uint4*)(x0 + kk + 16 + kA);
      a1.q[0] = *(const uint4*)(x1 + kk + kA);
      a1.q[1] = *(const uint4*)(x1 + kk + 16 + kA);
#pragma unroll
      for (int g = 0; g < NT; g += 4) {
        Frag16 b[4];
#pragma unroll
        for (int j = 0; j < 4; ++j) {
          const uint4* pb =
              (const uint4*)(lds + ((size_t)((g + j) * 16 + l16) * K + kk + kA));
          b[j].q[0] = pb[0];
          b[j].q[1] = pb[2];
        }
#pragma unroll
        for (int j = 0; j < 4; ++j) {
          acc[0][g + j] = __builtin_amdgcn_wmma_f32_16x16x32_f16(
              false, a0.v, false, b[j].v, (short)0, acc[0][g + j], false, false);
          acc[1][g + j] = __builtin_amdgcn_wmma_f32_16x16x32_f16(
              false, a1.v, false, b[j].v, (short)0, acc[1][g + j], false, false);
        }
      }
    }

    const int mb = hi * 8;        // C/D layout: lanes 16-31 hold M=8..15
#pragma unroll
    for (int half = 0; half < 2; ++half) {
#pragma unroll
      for (int t = 0; t < NT; ++t) {
        const int n  = t * 16 + l16;
        const float bv = bias[n];
#pragma unroll
        for (int r = 0; r < 8; ++r) {
          const int m = m0 + half * 16 + mb + r;
          if (m < M) {
            float y = acc[half][t][r] + bv;
            if (act == 1)      y = sp_half(y);
            else if (act == 2) y = sp_one(y);
            if (resid)  y += resid[(size_t)m * 128 + n];
            if (outF32) outF32[(size_t)m * 128 + n] = y;
            if (outH1)  outH1[(size_t)m * ldh1 + n] = (_Float16)y;
            if (outH2)  outH2[(size_t)m * ldh2 + n] = (_Float16)y;
          }
        }
      }
    }
  }
}

// Transpose fp32 W[K][Nout] -> f16 Wt[Nout][Kpad] (zero-pad K)
__global__ void k_transpose(const float* __restrict__ src, _Float16* __restrict__ dst,
                            int K, int Nout, int Kpad)
{
  int idx = blockIdx.x * blockDim.x + threadIdx.x;
  if (idx >= Nout * Kpad) return;
  int n = idx / Kpad, k = idx % Kpad;
  dst[idx] = (k < K) ? (_Float16)src[(size_t)k * Nout + n] : (_Float16)0.f;
}

// h0 = node_emb[atom_type]; fan out to fp32, f16, and hcat column block 0
__global__ void k_embed(const int* __restrict__ at, const float* __restrict__ emb,
                        float* __restrict__ hf, _Float16* __restrict__ hh,
                        _Float16* __restrict__ hcat, int N)
{
  int idx = blockIdx.x * blockDim.x + threadIdx.x;
  if (idx >= N * 128) return;
  int n = idx >> 7, c = idx & 127;
  float v = emb[(size_t)at[n] * 128 + c];
  hf[idx] = v;
  hh[idx] = (_Float16)v;
  hcat[(size_t)n * 512 + c] = (_Float16)v;
}

// RBF expansion -> f16 [E,32] (columns 30,31 zero padded)
__global__ void k_rbf(const float* __restrict__ dist, _Float16* __restrict__ rbf, int E)
{
  int idx = blockIdx.x * blockDim.x + threadIdx.x;
  if (idx >= E * 32) return;
  int e = idx >> 5, k = idx & 31;
  float v = 0.f;
  if (k < 30) {
    const float gap = 10.f / 29.f;
    float d = dist[e] - (float)k * gap;
    v = __expf(-d * d / gap);
  }
  rbf[idx] = (_Float16)v;
}

__global__ void k_zero(float* __restrict__ p, int n)
{
  int i = blockIdx.x * blockDim.x + threadIdx.x;
  if (i < n) p[i] = 0.f;
}

__global__ void k_cvt(const float* __restrict__ s, _Float16* __restrict__ d, int n)
{
  int i = blockIdx.x * blockDim.x + threadIdx.x;
  if (i < n) d[i] = (_Float16)s[i];
}

// msg = (h_new[src] + 1) * e ;  node[dst] += msg   (4 columns per thread)
__global__ void k_message(const float* __restrict__ hnew, const _Float16* __restrict__ e3,
                          const int* __restrict__ src, const int* __restrict__ dst,
                          float* __restrict__ node, int E)
{
  int idx = blockIdx.x * blockDim.x + threadIdx.x;
  if (idx >= E * 32) return;
  int e = idx >> 5;
  int c = (idx & 31) << 2;
  int s = src[e], d = dst[e];
  const float4 hv = *(const float4*)(hnew + (size_t)s * 128 + c);
  union { uint2 u; _Float16 h[4]; } ev;
  ev.u = *(const uint2*)(e3 + (size_t)e * 128 + c);
  float* np = node + (size_t)d * 128 + c;
  atomicAdd(np + 0, (hv.x + 1.f) * (float)ev.h[0]);
  atomicAdd(np + 1, (hv.y + 1.f) * (float)ev.h[1]);
  atomicAdd(np + 2, (hv.z + 1.f) * (float)ev.h[2]);
  atomicAdd(np + 3, (hv.w + 1.f) * (float)ev.h[3]);
}

// out[gid[n]] += r1[n,:] . W2 + b2
__global__ void k_readout(const _Float16* __restrict__ r1, const float* __restrict__ W2,
                          const float* __restrict__ b2, const int* __restrict__ gid,
                          float* __restrict__ out, int N)
{
  int n = blockIdx.x * blockDim.x + threadIdx.x;
  if (n >= N) return;
  const _Float16* rp = r1 + (size_t)n * 64;
  float s = b2[0];
#pragma unroll
  for (int j = 0; j < 64; ++j) s += (float)rp[j] * W2[j];
  atomicAdd(&out[gid[n]], s);
}

extern "C" void kernel_launch(void* const* d_in, const int* in_sizes, int n_in,
                              void* d_out, int out_size, void* d_ws, size_t ws_size,
                              hipStream_t stream)
{
  (void)n_in; (void)ws_size;
  const int N = in_sizes[0];
  const int E = in_sizes[1];
  const int B = out_size;
  const int L = 3;

  const int*   atom_type = (const int*)d_in[0];
  const int*   src  = (const int*)d_in[1];
  const int*   dst  = (const int*)d_in[2];
  const int*   gid  = (const int*)d_in[3];
  const float* dist = (const float*)d_in[4];
  const float* emb  = (const float*)d_in[5];
  const float* W_n1 = (const float*)d_in[6];  const float* b_n1 = (const float*)d_in[7];
  const float* W_c1 = (const float*)d_in[8];  const float* b_c1 = (const float*)d_in[9];
  const float* W_c2 = (const float*)d_in[10]; const float* b_c2 = (const float*)d_in[11];
  const float* W_c3 = (const float*)d_in[12]; const float* b_c3 = (const float*)d_in[13];
  const float* W_n2 = (const float*)d_in[14]; const float* b_n2 = (const float*)d_in[15];
  const float* W_n3 = (const float*)d_in[16]; const float* b_n3 = (const float*)d_in[17];
  const float* W_r1 = (const float*)d_in[18]; const float* b_r1 = (const float*)d_in[19];
  const float* W_r2 = (const float*)d_in[20]; const float* b_r2 = (const float*)d_in[21];
  float* out = (float*)d_out;

  // --- workspace partition (256B aligned) ---
  char* p = (char*)d_ws;
  auto alloc = [&](size_t bytes) -> char* {
    char* r = p; p += (bytes + 255) & ~(size_t)255; return r;
  };
  _Float16* wt_n1 = (_Float16*)alloc((size_t)L * 128 * 128 * 2);
  _Float16* wt_c1 = (_Float16*)alloc((size_t)L * 128 * 32 * 2);
  _Float16* wt_c2 = (_Float16*)alloc((size_t)L * 128 * 128 * 2);
  _Float16* wt_c3 = (_Float16*)alloc((size_t)L * 128 * 128 * 2);
  _Float16* wt_n2 = (_Float16*)alloc((size_t)L * 128 * 128 * 2);
  _Float16* wt_n3 = (_Float16*)alloc((size_t)L * 128 * 128 * 2);
  _Float16* wt_r1 = (_Float16*)alloc((size_t)64 * 512 * 2);
  float*    h_f32  = (float*)alloc((size_t)N * 128 * 4);
  float*    hnew   = (float*)alloc((size_t)N * 128 * 4);
  float*    node_f = (float*)alloc((size_t)N * 128 * 4);
  _Float16* h_h    = (_Float16*)alloc((size_t)N * 128 * 2);
  _Float16* node_h = (_Float16*)alloc((size_t)N * 128 * 2);
  _Float16* tmp_h  = (_Float16*)alloc((size_t)N * 128 * 2);
  _Float16* hcat   = (_Float16*)alloc((size_t)N * 512 * 2);
  _Float16* r1h    = (_Float16*)alloc((size_t)N * 64 * 2);
  _Float16* rbf    = (_Float16*)alloc((size_t)E * 32 * 2);
  _Float16* e1     = (_Float16*)alloc((size_t)E * 128 * 2);
  _Float16* e2     = (_Float16*)alloc((size_t)E * 128 * 2);

  // --- weight prep: transpose + f16 convert ---
  auto TP = [&](const float* s, _Float16* d, int K, int Nout, int Kpad) {
    int tot = Nout * Kpad;
    k_transpose<<<(tot + 255) / 256, 256, 0, stream>>>(s, d, K, Nout, Kpad);
  };
  for (int i = 0; i < L; ++i) {
    TP(W_n1 + (size_t)i * 128 * 128, wt_n1 + (size_t)i * 128 * 128, 128, 128, 128);
    TP(W_c1 + (size_t)i * 30 * 128,  wt_c1 + (size_t)i * 128 * 32,   30, 128,  32);
    TP(W_c2 + (size_t)i * 128 * 128, wt_c2 + (size_t)i * 128 * 128, 128, 128, 128);
    TP(W_c3 + (size_t)i * 128 * 128, wt_c3 + (size_t)i * 128 * 128, 128, 128, 128);
    TP(W_n2 + (size_t)i * 128 * 128, wt_n2 + (size_t)i * 128 * 128, 128, 128, 128);
    TP(W_n3 + (size_t)i * 128 * 128, wt_n3 + (size_t)i * 128 * 128, 128, 128, 128);
  }
  TP(W_r1, wt_r1, 512, 64, 512);

  k_embed<<<(N * 128 + 255) / 256, 256, 0, stream>>>(atom_type, emb, h_f32, h_h, hcat, N);
  k_rbf<<<(E * 32 + 255) / 256, 256, 0, stream>>>(dist, rbf, E);

  auto gemm8 = [&](const _Float16* X, int ldx, const _Float16* Wt, const float* bias,
                   int M, int K, int act, const float* resid, float* oF,
                   _Float16* oH1, int ldh1, _Float16* oH2, int ldh2) {
    int mtiles2 = (M + 31) / 32;
    int blocks = (mtiles2 + 3) / 4;
    if (blocks > 8192) blocks = 8192;
    size_t sh = (size_t)128 * K * 2;
    k_gemm<8><<<blocks, 128, sh, stream>>>(X, ldx, Wt, bias, M, K, act,
                                           resid, oF, oH1, ldh1, oH2, ldh2);
  };

  for (int i = 0; i < L; ++i) {
    const size_t wofs = (size_t)i * 128 * 128;
    // h_new = h @ W_n1 + b_n1
    gemm8(h_h, 128, wt_n1 + wofs, b_n1 + i * 128, N, 128, 0,
          nullptr, hnew, nullptr, 0, nullptr, 0);
    // edge MLP: e = sp(rbf@Wc1+b)@Wc2+b ; e = e@Wc3+b
    gemm8(rbf, 32, wt_c1 + (size_t)i * 128 * 32, b_c1 + i * 128, E, 32, 1,
          nullptr, nullptr, e1, 128, nullptr, 0);
    gemm8(e1, 128, wt_c2 + wofs, b_c2 + i * 128, E, 128, 0,
          nullptr, nullptr, e2, 128, nullptr, 0);
    gemm8(e2, 128, wt_c3 + wofs, b_c3 + i * 128, E, 128, 0,
          nullptr, nullptr, e1, 128, nullptr, 0);   // e3 reuses e1
    // scatter: node = segsum((h_new[src]+1)*e, dst)
    k_zero<<<(N * 128 + 255) / 256, 256, 0, stream>>>(node_f, N * 128);
    k_message<<<(E * 32 + 255) / 256, 256, 0, stream>>>(hnew, e1, src, dst, node_f, E);
    k_cvt<<<(N * 128 + 255) / 256, 256, 0, stream>>>(node_f, node_h, N * 128);
    // new_node = sp(node@Wn2+b)@Wn3+b ; h += new_node ; also h_f16, hcat block
    gemm8(node_h, 128, wt_n2 + wofs, b_n2 + i * 128, N, 128, 1,
          nullptr, nullptr, tmp_h, 128, nullptr, 0);
    gemm8(tmp_h, 128, wt_n3 + wofs, b_n3 + i * 128, N, 128, 0,
          h_f32, h_f32, h_h, 128, hcat + (size_t)(i + 1) * 128, 512);
  }

  // readout: r1 = sp1(hcat @ W_r1 + b_r1)  [N,64]
  {
    int mtiles2 = (N + 31) / 32;
    int blocks = (mtiles2 + 3) / 4;
    if (blocks > 8192) blocks = 8192;
    k_gemm<4><<<blocks, 128, (size_t)64 * 512 * 2, stream>>>(
        hcat, 512, wt_r1, b_r1, N, 512, 2,
        nullptr, nullptr, r1h, 64, nullptr, 0);
  }
  k_zero<<<(B + 255) / 256, 256, 0, stream>>>(out, B);
  k_readout<<<(N + 255) / 256, 256, 0, stream>>>(r1h, W_r2, b_r2, gid, out, N);
}
